// GNN_996432412877
// MI455X (gfx1250) — compile-verified, matrix operand order checked by reading, not run on hardware
//
#include <hip/hip_runtime.h>

typedef __attribute__((ext_vector_type(16))) _Float16 v16h;
typedef __attribute__((ext_vector_type(8)))  _Float16 v8h;
typedef __attribute__((ext_vector_type(8)))  float    v8f;
typedef __attribute__((ext_vector_type(4)))  float    f32x4;

#define IN_SIZE   64
#define HIDDEN    128
#define WAVES     8
#define TILE_M    16
#define BLOCK_EDGES (WAVES * TILE_M)

// padded LDS leading dims (in halves) to avoid bank conflicts on B128 loads
#define W1_LD 72    // 64  + 8  -> row stride 144B,  (36n)%64 distinct for n mod 16
#define W2_LD 136   // 128 + 8  -> row stride 272B,  (68n)%64 = (4n)%64 distinct
#define H_LD  136

// Hardware tanh (CDNA5 TRANS op, co-executes with WMMA); fallback to libm.
#if defined(__has_builtin)
#  if __has_builtin(__builtin_amdgcn_tanhf)
#    define FAST_TANH(x) __builtin_amdgcn_tanhf(x)
#  endif
#endif
#ifndef FAST_TANH
#  define FAST_TANH(x) tanhf(x)
#endif

struct __align__(16) Smem {
  _Float16 w1[HIDDEN * W1_LD];            // 18432 B
  _Float16 w2[HIDDEN * W2_LD];            // 34816 B
  _Float16 w3[HIDDEN * W2_LD];            // 34816 B
  _Float16 h [WAVES * TILE_M * H_LD];     // 34816 B (per-wave 16x128 activation tile)
  float b1[HIDDEN], b2[HIDDEN], b3[HIDDEN], w4[HIDDEN];
  float b4;
};

// Load a 16x32 f16 fragment (A or B) from LDS per the CDNA5 16-bit WMMA layout:
// lane L holds halves h=0..7 -> K = koff..koff+7, h=8..15 -> K = koff+16..koff+23,
// with koff = kstep*32 + (L>=16 ? 8 : 0).  'row' is the matrix row (M for A, N for B).
__device__ __forceinline__ v16h lds_frag(const _Float16* base, int row, int ld,
                                         int kstep, int lane) {
  const int koff = kstep * 32 + ((lane >> 4) << 3);
  const _Float16* p = base + row * ld + koff;
  v8h lo = *(const v8h*)(p);
  v8h hi = *(const v8h*)(p + 16);
  return __builtin_shufflevector(lo, hi, 0,1,2,3,4,5,6,7,8,9,10,11,12,13,14,15);
}

// Build an A fragment for one 16-edge row tile from global f32 edge_attr.
__device__ __forceinline__ v16h a_frag_global(const float* __restrict__ row,
                                              int kstep, int lane) {
  const int koff = kstep * 32 + ((lane >> 4) << 3);
  f32x4 a0 = *(const f32x4*)(row + koff + 0);
  f32x4 a1 = *(const f32x4*)(row + koff + 4);
  f32x4 a2 = *(const f32x4*)(row + koff + 16);
  f32x4 a3 = *(const f32x4*)(row + koff + 20);
  v16h a;
#pragma unroll
  for (int i = 0; i < 4; ++i) {
    a[i]      = (_Float16)a0[i];
    a[4 + i]  = (_Float16)a1[i];
    a[8 + i]  = (_Float16)a2[i];
    a[12 + i] = (_Float16)a3[i];
  }
  return a;
}

#define WMMA_F16(A, B, C) \
  __builtin_amdgcn_wmma_f32_16x16x32_f16(false, (A), false, (B), (short)0, (C), false, false)

__global__ __launch_bounds__(256) void gnn_edge_mlp(
    const float* __restrict__ edge_attr,
    const float* __restrict__ W1, const float* __restrict__ b1,
    const float* __restrict__ W2, const float* __restrict__ b2,
    const float* __restrict__ W3, const float* __restrict__ b3,
    const float* __restrict__ W4, const float* __restrict__ b4,
    float* __restrict__ out, int E, int nTiles) {
  __shared__ Smem sm;
  const int tid  = threadIdx.x;
  const int lane = tid & 31;
  const int wave = tid >> 5;
  const int m    = lane & 15;

  // ---- stage weights (f32 -> f16) and biases into LDS, once per block ----
  for (int i = tid; i < HIDDEN * IN_SIZE; i += 256)
    sm.w1[(i >> 6) * W1_LD + (i & 63)] = (_Float16)W1[i];
  for (int i = tid; i < HIDDEN * HIDDEN; i += 256) {
    const int r = i >> 7, c = i & 127;
    sm.w2[r * W2_LD + c] = (_Float16)W2[i];
    sm.w3[r * W2_LD + c] = (_Float16)W3[i];
  }
  for (int i = tid; i < HIDDEN; i += 256) {
    sm.b1[i] = b1[i]; sm.b2[i] = b2[i]; sm.b3[i] = b3[i]; sm.w4[i] = W4[i];
  }
  __syncthreads();

  // hoist per-lane invariants out of the tile loop (D layout: N = nt*16 + m)
  float bv1[8], bv2[8], bv3[8], w4v[8];
#pragma unroll
  for (int nt = 0; nt < 8; ++nt) {
    bv1[nt] = sm.b1[nt * 16 + m];
    bv2[nt] = sm.b2[nt * 16 + m];
    bv3[nt] = sm.b3[nt * 16 + m];
    w4v[nt] = sm.w4[nt * 16 + m];
  }
  const float b4r = b4[0];

  _Float16* hb = &sm.h[wave * TILE_M * H_LD];
  const int rowoff = (lane >> 4) << 3;   // D layout: lanes 16-31 hold rows M+8

  for (int tb = blockIdx.x; tb < nTiles; tb += gridDim.x) {
    const int e0 = tb * BLOCK_EDGES + wave * TILE_M;
    int er = e0 + m; if (er >= E) er = E - 1;            // clamp (E % 128 == 0 anyway)
    const float* arow = edge_attr + (size_t)er * IN_SIZE;

    // prefetch next tile's rows into the cache hierarchy
    {
      const long en = (long)(tb + gridDim.x) * BLOCK_EDGES + wave * TILE_M + m;
      if (en < E) __builtin_prefetch(edge_attr + (size_t)en * IN_SIZE, 0, 1);
    }

    v8f acc[8];

    // ---------------- layer 1: [16x64] x [64x128] ----------------
#pragma unroll
    for (int nt = 0; nt < 8; ++nt) {
#pragma unroll
      for (int r = 0; r < 8; ++r) acc[nt][r] = bv1[nt];
    }
#pragma unroll
    for (int ks = 0; ks < 2; ++ks) {
      const v16h A = a_frag_global(arow, ks, lane);
#pragma unroll
      for (int nt = 0; nt < 8; ++nt) {
        const v16h B = lds_frag(sm.w1, nt * 16 + m, W1_LD, ks, lane);
        acc[nt] = WMMA_F16(A, B, acc[nt]);
      }
    }
#pragma unroll
    for (int nt = 0; nt < 8; ++nt) {
      const int n = nt * 16 + m;
#pragma unroll
      for (int r = 0; r < 8; ++r)
        hb[(r + rowoff) * H_LD + n] = (_Float16)FAST_TANH(acc[nt][r]);
    }

    // ---------------- layer 2: [16x128] x [128x128] ----------------
#pragma unroll
    for (int nt = 0; nt < 8; ++nt) {
#pragma unroll
      for (int r = 0; r < 8; ++r) acc[nt][r] = bv2[nt];
    }
#pragma unroll
    for (int ks = 0; ks < 4; ++ks) {
      const v16h A = lds_frag(hb, m, H_LD, ks, lane);
#pragma unroll
      for (int nt = 0; nt < 8; ++nt) {
        const v16h B = lds_frag(sm.w2, nt * 16 + m, W2_LD, ks, lane);
        acc[nt] = WMMA_F16(A, B, acc[nt]);
      }
    }
#pragma unroll
    for (int nt = 0; nt < 8; ++nt) {
      const int n = nt * 16 + m;
#pragma unroll
      for (int r = 0; r < 8; ++r)
        hb[(r + rowoff) * H_LD + n] = (_Float16)FAST_TANH(acc[nt][r]);
    }

    // ---------------- layer 3 + layer 4 fused ----------------
    // gemm with W3, then psum[r] = sum_n tanh(h3[row,n]) * W4[n] without an
    // LDS round-trip: tanh stays in D-layout registers, dot is accumulated
    // per lane and reduced across each 16-lane half with shfl_xor.
#pragma unroll
    for (int nt = 0; nt < 8; ++nt) {
#pragma unroll
      for (int r = 0; r < 8; ++r) acc[nt][r] = bv3[nt];
    }
#pragma unroll
    for (int ks = 0; ks < 4; ++ks) {
      const v16h A = lds_frag(hb, m, H_LD, ks, lane);
#pragma unroll
      for (int nt = 0; nt < 8; ++nt) {
        const v16h B = lds_frag(sm.w3, nt * 16 + m, W2_LD, ks, lane);
        acc[nt] = WMMA_F16(A, B, acc[nt]);
      }
    }
    float psum[8];
#pragma unroll
    for (int r = 0; r < 8; ++r) psum[r] = 0.0f;
#pragma unroll
    for (int nt = 0; nt < 8; ++nt) {
#pragma unroll
      for (int r = 0; r < 8; ++r)
        psum[r] += FAST_TANH(acc[nt][r]) * w4v[nt];
    }
    // reduce over the 16 lanes holding one row (lanes 0-15: rows 0-7,
    // lanes 16-31: rows 8-15)
#pragma unroll
    for (int off = 1; off < 16; off <<= 1) {
#pragma unroll
      for (int r = 0; r < 8; ++r)
        psum[r] += __shfl_xor(psum[r], off, 16);
    }
    if (m == 0) {
#pragma unroll
      for (int r = 0; r < 8; ++r) {
        const int e = e0 + r + rowoff;
        if (e < E)
          out[e] = 0.5f + 0.5f * FAST_TANH(0.5f * (psum[r] + b4r));  // sigmoid
      }
    }
  }
}

extern "C" void kernel_launch(void* const* d_in, const int* in_sizes, int n_in,
                              void* d_out, int out_size, void* d_ws, size_t ws_size,
                              hipStream_t stream) {
  // inputs: 0=x, 1=edge_index(int64, unused), 2=edge_attr, 3=W1, 4=b1,
  //         5=W2, 6=b2, 7=W3, 8=b3, 9=W4, 10=b4
  const float* edge_attr = (const float*)d_in[2];
  const float* W1 = (const float*)d_in[3];
  const float* b1 = (const float*)d_in[4];
  const float* W2 = (const float*)d_in[5];
  const float* b2 = (const float*)d_in[6];
  const float* W3 = (const float*)d_in[7];
  const float* b3 = (const float*)d_in[8];
  const float* W4 = (const float*)d_in[9];
  const float* b4 = (const float*)d_in[10];
  float* out = (float*)d_out;

  const int E = in_sizes[2] / IN_SIZE;
  const int nTiles = (E + BLOCK_EDGES - 1) / BLOCK_EDGES;
  const int grid = nTiles < 768 ? nTiles : 768;

  gnn_edge_mlp<<<dim3(grid), dim3(256), 0, stream>>>(
      edge_attr, W1, b1, W2, b2, W3, b3, W4, b4, out, E, nTiles);
}
